// lstm_from_scratch_8907762172253
// MI455X (gfx1250) — matrix operand order repaired and checked by reading, not compiled, not run on hardware
//
#include <hip/hip_runtime.h>
#include <hip/hip_bf16.h>

// ---------------------------------------------------------------------------
// LSTM forward on gfx1250 (CDNA5): bf16 WMMA (V_WMMA_F32_16X16X32_BF16),
// fp32 accumulate / cell state.
//   Phase 1: big parallel GEMM x@W -> x_proj   (M=32768, K=1024, N=4096)
//   Phase 2: 512 per-step launches h@U + x_proj, fused gate math.
// Weight tiles staged to LDS with GLOBAL_LOAD_ASYNC_TO_LDS_B128 (ASYNCcnt).
// ---------------------------------------------------------------------------

#define SEQ 512
#define BATCH 64
#define ISZ 1024
#define HSZ 1024
#define N4 4096
#define M1 (SEQ * BATCH)   // 32768

typedef __attribute__((ext_vector_type(16))) __bf16 v16bf;
typedef __attribute__((ext_vector_type(8)))  float  v8f;

union FragU { unsigned int u[8]; v16bf v; };

__device__ __forceinline__ unsigned short f32_to_bf16(float f) {
  unsigned int u = __builtin_bit_cast(unsigned int, f);
  unsigned int lsb = (u >> 16) & 1u;
  u += 0x7FFFu + lsb;               // round to nearest even
  return (unsigned short)(u >> 16);
}

__device__ __forceinline__ float sigm(float x) { return 1.0f / (1.0f + __expf(-x)); }

// Async global->LDS copy, 16 bytes per lane. Tracked with ASYNCcnt.
__device__ __forceinline__ void async_copy_b128(void* lds_ptr, const void* gptr) {
  unsigned lds_addr = (unsigned)(uintptr_t)lds_ptr;           // low 32b = LDS offset
  unsigned long long ga = (unsigned long long)(uintptr_t)gptr;
  asm volatile("global_load_async_to_lds_b128 %0, %1, off"
               :: "v"(lds_addr), "v"(ga) : "memory");
}
__device__ __forceinline__ void wait_async0() {
  asm volatile("s_wait_asynccnt 0" ::: "memory");
}

// A fragment (16x32 bf16, M x K), per ISA layout:
//   lane L (L<16): M=L, K pairs {0,2,4,6} in v0..3, {16,18,20,22} in v4..7
//   lane L+16:     M=L, K pairs {8..14} in v0..3, {24..30} in v4..7
__device__ __forceinline__ v16bf frag_a(const unsigned short* T, int stride,
                                        int row_base, int lane) {
  const int m = row_base + (lane & 15);
  const int half = lane >> 4;
  FragU f;
#pragma unroll
  for (int v = 0; v < 8; ++v) {
    int kk = (v < 4) ? (half * 8 + v * 2) : (16 + half * 8 + (v - 4) * 2);
    f.u[v] = *(const unsigned int*)(T + m * stride + kk);
  }
  return f.v;
}

// B fragment (32x16 bf16, K x N), LDS tile stored K-contiguous: T[n][k].
//   lane L (L<16): N=L, K 0..7,16..23 ; lane L+16: N=L, K 8..15,24..31
__device__ __forceinline__ v16bf frag_b(const unsigned short* T, int stride,
                                        int col_base, int lane) {
  const int n = col_base + (lane & 15);
  const int half = lane >> 4;
  FragU f;
#pragma unroll
  for (int v = 0; v < 8; ++v) {
    int kk = (v < 4) ? (half * 8 + v * 2) : (16 + half * 8 + (v - 4) * 2);
    f.u[v] = *(const unsigned int*)(T + n * stride + kk);
  }
  return f.v;
}

// ---------------------------------------------------------------------------
// Pack weights TRANSPOSED: Wt[n][k] (n = gate-concat column, 0..4095), bf16.
// Makes every GEMM B-tile row a contiguous 64-byte run -> async B128 copies.
// ---------------------------------------------------------------------------
__global__ void lstm_pack_params(
    const float* __restrict__ Wi, const float* __restrict__ Ui, const float* __restrict__ bi,
    const float* __restrict__ Wf, const float* __restrict__ Uf, const float* __restrict__ bf_,
    const float* __restrict__ Wc, const float* __restrict__ Uc, const float* __restrict__ bc,
    const float* __restrict__ Wo, const float* __restrict__ Uo, const float* __restrict__ bo,
    unsigned short* __restrict__ Wt, unsigned short* __restrict__ Ut,
    float* __restrict__ bcat) {
  const int idx = blockIdx.x * blockDim.x + threadIdx.x;   // 0 .. 1024*1024-1
  const int k = idx >> 10;
  const int h = idx & 1023;
  const float* Ws[4] = {Wi, Wf, Wc, Wo};
  const float* Us[4] = {Ui, Uf, Uc, Uo};
#pragma unroll
  for (int g = 0; g < 4; ++g) {
    Wt[(size_t)(g * HSZ + h) * ISZ + k] = f32_to_bf16(Ws[g][(size_t)k * HSZ + h]);
    Ut[(size_t)(g * HSZ + h) * HSZ + k] = f32_to_bf16(Us[g][(size_t)k * HSZ + h]);
  }
  if (idx < N4) {
    const float* bs[4] = {bi, bf_, bc, bo};
    bcat[idx] = bs[idx >> 10][idx & 1023];
  }
}

__global__ void lstm_init_state(const float* __restrict__ h0, const float* __restrict__ c0,
                                float* __restrict__ hbuf, float* __restrict__ cbuf) {
  const int idx = blockIdx.x * blockDim.x + threadIdx.x;   // 0 .. 65535
  hbuf[idx] = h0[idx];
  cbuf[idx] = c0[idx];
}

// ---------------------------------------------------------------------------
// Phase 1 GEMM: x_proj[M1,4096] = bf16(x[M1,1024]) @ W + bcat
// Block: 128x128 C-tile, 256 threads (8 waves), wave = 64x32 sub-tile.
// W tile staged via async B128 copies; x tile converted f32->bf16 by VALU.
// ---------------------------------------------------------------------------
__global__ void __launch_bounds__(256)
lstm_xproj_gemm(const float* __restrict__ x, const unsigned short* __restrict__ Wt,
                const float* __restrict__ bcat, float* __restrict__ xproj) {
  __shared__ unsigned short Ax[128][40];   // x tile, bf16, [m][k]
  __shared__ unsigned short BT[128][40];   // W tile, bf16, [n][k]

  const int tid  = threadIdx.x;
  const int lane = tid & 31;
  const int w    = tid >> 5;
  const int half = lane >> 4;
  const int m0 = blockIdx.y * 128;
  const int n0 = blockIdx.x * 128;
  const int wm = (w >> 2) * 64;
  const int wn = (w & 3) * 32;

  v8f acc[4][2];
#pragma unroll
  for (int i = 0; i < 4; ++i)
#pragma unroll
    for (int j = 0; j < 2; ++j) acc[i][j] = 0.0f;

  for (int k0 = 0; k0 < ISZ; k0 += 32) {
    // async-stage B: 128 rows x 64B = 512 x B128, 2 per thread
#pragma unroll
    for (int i = 0; i < 2; ++i) {
      int idx = tid + i * 256;
      int row = idx >> 2, seg = idx & 3;
      async_copy_b128(&BT[row][seg * 8],
                      Wt + (size_t)(n0 + row) * ISZ + k0 + seg * 8);
    }
    // overlap: stage A (128x32) f32 -> bf16
#pragma unroll
    for (int i = 0; i < 16; ++i) {
      int idx = tid + i * 256;
      int r = idx >> 5, c = idx & 31;
      Ax[r][c] = f32_to_bf16(x[(size_t)(m0 + r) * ISZ + k0 + c]);
    }
    wait_async0();
    __syncthreads();

    v16bf a[4], b[2];
#pragma unroll
    for (int fi = 0; fi < 4; ++fi) a[fi] = frag_a(&Ax[0][0], 40, wm + fi * 16, lane);
#pragma unroll
    for (int fj = 0; fj < 2; ++fj) b[fj] = frag_b(&BT[0][0], 40, wn + fj * 16, lane);
#pragma unroll
    for (int fi = 0; fi < 4; ++fi)
#pragma unroll
      for (int fj = 0; fj < 2; ++fj)
        acc[fi][fj] = __builtin_amdgcn_wmma_f32_16x16x32_bf16(
            false, a[fi], false, b[fj], (short)0, acc[fi][fj], false, false);
    __syncthreads();
  }

  // store C (+bias): VGPR r -> M = r + 8*half, N = lane&15
#pragma unroll
  for (int fi = 0; fi < 4; ++fi) {
#pragma unroll
    for (int fj = 0; fj < 2; ++fj) {
      int n = n0 + wn + fj * 16 + (lane & 15);
      float bv = bcat[n];
#pragma unroll
      for (int r = 0; r < 8; ++r) {
        int m = m0 + wm + fi * 16 + half * 8 + r;
        xproj[(size_t)m * N4 + n] = acc[fi][fj][r] + bv;
      }
    }
  }
}

// ---------------------------------------------------------------------------
// Phase 2 step: gates = x_proj[t] + h_in @ U ; fused LSTM cell update.
// Grid: 16 blocks x 64 h-columns. Each wave owns 2 (m,n) fragment positions
// and all 4 gate accumulators -> elementwise stays entirely in registers.
// ---------------------------------------------------------------------------
__global__ void __launch_bounds__(256)
lstm_step(const float* __restrict__ h_in, float* __restrict__ h_out,
          float* __restrict__ c_buf, const unsigned short* __restrict__ Ut,
          const float* __restrict__ xproj, float* __restrict__ out_h,
          float* __restrict__ out_hT, float* __restrict__ out_cT, int t) {
  __shared__ unsigned short Ah[64][40];    // h tile bf16 [b][k]
  __shared__ unsigned short BT[256][40];   // U tile bf16 [j][k], j = g*64+jj

  const int tid  = threadIdx.x;
  const int lane = tid & 31;
  const int w    = tid >> 5;
  const int half = lane >> 4;
  const int n0 = blockIdx.x * 64;          // h-column block
  const int fi = w & 3;                    // M fragment row (16 rows)
  const int fjbase = (w >> 2) * 32;        // 2 N fragments of 16

  v8f acc[4][2];
#pragma unroll
  for (int g = 0; g < 4; ++g)
#pragma unroll
    for (int j = 0; j < 2; ++j) acc[g][j] = 0.0f;

  for (int k0 = 0; k0 < HSZ; k0 += 32) {
    // async-stage U: 256 rows x 64B = 1024 x B128, 4 per thread
#pragma unroll
    for (int i = 0; i < 4; ++i) {
      int idx = tid + i * 256;
      int j = idx >> 2, seg = idx & 3;
      int g = j >> 6, jj = j & 63;
      async_copy_b128(&BT[j][seg * 8],
                      Ut + (size_t)(g * HSZ + n0 + jj) * HSZ + k0 + seg * 8);
    }
    // overlap: stage h (64x32) f32 -> bf16
#pragma unroll
    for (int i = 0; i < 8; ++i) {
      int idx = tid + i * 256;
      int r = idx >> 5, c = idx & 31;
      Ah[r][c] = f32_to_bf16(h_in[(size_t)r * HSZ + k0 + c]);
    }
    wait_async0();
    __syncthreads();

    v16bf a = frag_a(&Ah[0][0], 40, fi * 16, lane);
#pragma unroll
    for (int g = 0; g < 4; ++g) {
#pragma unroll
      for (int j = 0; j < 2; ++j) {
        v16bf b = frag_b(&BT[0][0], 40, g * 64 + fjbase + j * 16, lane);
        acc[g][j] = __builtin_amdgcn_wmma_f32_16x16x32_bf16(
            false, a, false, b, (short)0, acc[g][j], false, false);
      }
    }
    __syncthreads();
  }

  // fused gate math, all in registers (i,f,g,o for a (b,n) live in same lane)
#pragma unroll
  for (int j = 0; j < 2; ++j) {
    int ncol = n0 + fjbase + j * 16 + (lane & 15);
#pragma unroll
    for (int r = 0; r < 8; ++r) {
      int b = fi * 16 + half * 8 + r;                 // batch row
      size_t xp = (size_t)(t * BATCH + b) * N4;
      float gi = sigm(acc[0][j][r] + xproj[xp + 0 * HSZ + ncol]);
      float gf = sigm(acc[1][j][r] + xproj[xp + 1 * HSZ + ncol]);
      float gg = tanhf(acc[2][j][r] + xproj[xp + 2 * HSZ + ncol]);
      float go = sigm(acc[3][j][r] + xproj[xp + 3 * HSZ + ncol]);
      size_t ci = (size_t)b * HSZ + ncol;
      float cn = gf * c_buf[ci] + gi * gg;
      float hn = go * tanhf(cn);
      c_buf[ci] = cn;
      h_out[ci] = hn;
      out_h[(size_t)b * SEQ * HSZ + (size_t)t * HSZ + ncol] = hn;
      if (t == SEQ - 1) { out_hT[ci] = hn; out_cT[ci] = cn; }
    }
  }
}

// ---------------------------------------------------------------------------
extern "C" void kernel_launch(void* const* d_in, const int* in_sizes, int n_in,
                              void* d_out, int out_size, void* d_ws, size_t ws_size,
                              hipStream_t stream) {
  (void)in_sizes; (void)n_in; (void)out_size; (void)ws_size;
  const float* x  = (const float*)d_in[0];
  const float* h0 = (const float*)d_in[1];
  const float* c0 = (const float*)d_in[2];
  const float* Wi = (const float*)d_in[3];  const float* Ui = (const float*)d_in[4];
  const float* bi = (const float*)d_in[5];
  const float* Wf = (const float*)d_in[6];  const float* Uf = (const float*)d_in[7];
  const float* bf_ = (const float*)d_in[8];
  const float* Wc = (const float*)d_in[9];  const float* Uc = (const float*)d_in[10];
  const float* bc = (const float*)d_in[11];
  const float* Wo = (const float*)d_in[12]; const float* Uo = (const float*)d_in[13];
  const float* bo = (const float*)d_in[14];

  float* out_h  = (float*)d_out;                          // [B, S, H]
  float* out_hT = out_h + (size_t)BATCH * SEQ * HSZ;      // [B, H]
  float* out_cT = out_hT + (size_t)BATCH * HSZ;           // [B, H]

  char* p = (char*)d_ws;
  auto take = [&](size_t bytes) { char* r = p; p += (bytes + 255) & ~(size_t)255; return r; };
  unsigned short* Wt = (unsigned short*)take((size_t)ISZ * N4 * 2);  // [4096][1024]
  unsigned short* Ut = (unsigned short*)take((size_t)HSZ * N4 * 2);  // [4096][1024]
  float* bcat  = (float*)take((size_t)N4 * 4);
  float* xproj = (float*)take((size_t)M1 * N4 * 4);
  float* hb0   = (float*)take((size_t)BATCH * HSZ * 4);
  float* hb1   = (float*)take((size_t)BATCH * HSZ * 4);
  float* cb    = (float*)take((size_t)BATCH * HSZ * 4);

  lstm_pack_params<<<(ISZ * HSZ) / 256, 256, 0, stream>>>(
      Wi, Ui, bi, Wf, Uf, bf_, Wc, Uc, bc, Wo, Uo, bo, Wt, Ut, bcat);
  lstm_init_state<<<(BATCH * HSZ) / 256, 256, 0, stream>>>(h0, c0, hb0, cb);

  dim3 gg(N4 / 128, M1 / 128);   // (32, 256)
  lstm_xproj_gemm<<<gg, 256, 0, stream>>>(x, Wt, bcat, xproj);

  for (int t = 0; t < SEQ; ++t) {
    float* hin  = (t & 1) ? hb1 : hb0;
    float* hout = (t & 1) ? hb0 : hb1;
    lstm_step<<<HSZ / 64, 256, 0, stream>>>(hin, hout, cb, Ut, xproj,
                                            out_h, out_hT, out_cT, t);
  }
}
